// LSTM_31645319037176
// MI455X (gfx1250) — matrix operand order repaired and checked by reading, not compiled, not run on hardware
//
#include <hip/hip_runtime.h>
#include <hip/hip_bf16.h>

// ---------------------------------------------------------------------------
// LSTM on MI455X (gfx1250): bf16 WMMA recurrent GEMMs, fused gate/cell update.
//   gate0 = Wh/bh (tanh -> cdash), gate1 = Wu/bu (sigmoid -> fu),
//   gate2 = Wf/bf (sigmoid -> ff).  Wo/bo are dead in the reference.
// Wave tile: 16 rows x 32 cols x 3 gates -> 256 waves/step (32 blocks),
// halving the serial WMMA chain per step vs a 16x64 tile.
// ---------------------------------------------------------------------------

typedef __attribute__((ext_vector_type(16))) __bf16 v16bf;
typedef __attribute__((ext_vector_type(8)))  float  v8f;
typedef unsigned short ushort_t;

static constexpr int B_   = 128;
static constexpr int T_   = 256;
static constexpr int IN_  = 512;
static constexpr int HID_ = 1024;
static constexpr int OUT_ = 512;
static constexpr int KD_  = IN_ + HID_;   // 1536

__device__ __forceinline__ ushort_t f2bf(float f) {
    unsigned u = __float_as_uint(f);
    unsigned r = (u + 0x7FFFu + ((u >> 16) & 1u)) >> 16;   // RNE
    return (ushort_t)r;
}
__device__ __forceinline__ float bf2f(ushort_t h) {
    return __uint_as_float(((unsigned)h) << 16);
}
__device__ __forceinline__ float sigmoidf(float x) {
    return 1.0f / (1.0f + __expf(-x));
}

union FragBF { uint4 u[2]; v16bf v; };

// -------- pack fp32 weights [K][N] -> bf16 transposed [gate][N][K] ----------
__global__ void k_pack_w(const float* __restrict__ Wh,
                         const float* __restrict__ Wu,
                         const float* __restrict__ Wf,
                         ushort_t* __restrict__ wt) {
    size_t i = (size_t)blockIdx.x * blockDim.x + threadIdx.x;
    size_t per = (size_t)HID_ * KD_;
    if (i >= 3 * per) return;
    int g = (int)(i / per);
    size_t r = i % per;
    int n = (int)(r / KD_);
    int k = (int)(r % KD_);
    const float* W = (g == 0) ? Wh : (g == 1) ? Wu : Wf;
    wt[i] = f2bf(W[(size_t)k * HID_ + n]);
}

// -------- x fp32 -> bf16 (same [B][T][IN] layout) ---------------------------
__global__ void k_xconv(const float* __restrict__ x, ushort_t* __restrict__ xb,
                        size_t n) {
    size_t i = (size_t)blockIdx.x * blockDim.x + threadIdx.x;
    if (i < n) xb[i] = f2bf(x[i]);
}

// -------- zero a0 (bf16) and c0 (f32) ---------------------------------------
__global__ void k_init(ushort_t* __restrict__ a0, float* __restrict__ c0) {
    int i = blockIdx.x * blockDim.x + threadIdx.x;
    if (i < B_ * HID_) { a0[i] = 0; c0[i] = 0.0f; }
}

// -------- one recurrent step: 3 gate GEMMs + cell update, fully fused -------
// wave tile: 16 rows x 32 cols x 3 gates. 8 m-tiles * 32 n32-tiles = 256 waves.
__global__ void __launch_bounds__(256)
k_step(const ushort_t* __restrict__ xbf,     // [B][T][IN] bf16
       const ushort_t* __restrict__ aprev,   // [B][HID] bf16
       ushort_t* __restrict__ anext,         // [B][HID] bf16
       float* __restrict__ cst,              // [B][HID] f32 (in/out)
       const ushort_t* __restrict__ wt,      // [3][HID][KD] bf16 (W^T)
       const float* __restrict__ bh,
       const float* __restrict__ bu,
       const float* __restrict__ bff,
       int t) {
    const int lane   = threadIdx.x & 31;
    const int waveId = blockIdx.x * (blockDim.x >> 5) + (threadIdx.x >> 5);
    const int mb     = waveId >> 5;          // 0..7
    const int n32    = waveId & 31;          // 0..31
    const int m_base = mb << 4;
    const int n_base = n32 << 5;

    const int hi     = lane >> 4;            // 0 or 1
    const int khalfA = hi << 3;              // A frag: +0 / +8
    const int khalfB = hi << 3;              // B frag: +0 / +8
    const int row    = m_base + (lane & 15); // A row (batch index)
    const int ncol   = lane & 15;            // B/D column within tile

    const ushort_t* xrow = xbf + ((size_t)row * T_ + t) * IN_;
    const ushort_t* arow = aprev + (size_t)row * HID_;
    // weight row base for this wave's two n-tiles, per gate
    const ushort_t* wbase = wt + (size_t)(n_base + ncol) * KD_ + khalfB;

    v8f acc[3][2];
#pragma unroll
    for (int g = 0; g < 3; ++g)
#pragma unroll
        for (int n = 0; n < 2; ++n) acc[g][n] = (v8f)(0.0f);

    // ---- phase 1: k over x part of comb (k in [0, IN)) ----
    for (int kb = 0; kb < IN_; kb += 32) {
        FragBF fa;
        fa.u[0] = *(const uint4*)(xrow + kb + khalfA);
        fa.u[1] = *(const uint4*)(xrow + kb + 16 + khalfA);
#pragma unroll
        for (int g = 0; g < 3; ++g) {
#pragma unroll
            for (int nt = 0; nt < 2; ++nt) {
                const ushort_t* wr = wbase + (size_t)g * HID_ * KD_ +
                                     (size_t)(nt << 4) * KD_ + kb;
                FragBF fb;
                fb.u[0] = *(const uint4*)(wr);
                fb.u[1] = *(const uint4*)(wr + 16);
                acc[g][nt] = __builtin_amdgcn_wmma_f32_16x16x32_bf16(
                    false, fa.v, false, fb.v, (short)0, acc[g][nt], false, false);
            }
        }
    }

    // ---- phase 2: k over a_prev part of comb (k in [IN, KD)) ----
    for (int kb = 0; kb < HID_; kb += 32) {
        FragBF fa;
        fa.u[0] = *(const uint4*)(arow + kb + khalfA);
        fa.u[1] = *(const uint4*)(arow + kb + 16 + khalfA);
#pragma unroll
        for (int g = 0; g < 3; ++g) {
#pragma unroll
            for (int nt = 0; nt < 2; ++nt) {
                const ushort_t* wr = wbase + (size_t)g * HID_ * KD_ +
                                     (size_t)(nt << 4) * KD_ + IN_ + kb;
                FragBF fb;
                fb.u[0] = *(const uint4*)(wr);
                fb.u[1] = *(const uint4*)(wr + 16);
                acc[g][nt] = __builtin_amdgcn_wmma_f32_16x16x32_bf16(
                    false, fa.v, false, fb.v, (short)0, acc[g][nt], false, false);
            }
        }
    }

    // Epilogue: identical C/D layouts across gates -> elementwise combine.
#pragma unroll
    for (int nt = 0; nt < 2; ++nt) {
        const int n = n_base + (nt << 4) + ncol;
        const float vbh = bh[n], vbu = bu[n], vbf = bff[n];
#pragma unroll
        for (int r = 0; r < 8; ++r) {
            const int m = m_base + r + (hi << 3);
            const size_t idx = (size_t)m * HID_ + n;
            const float cdash = tanhf(acc[0][nt][r] + vbh);
            const float fu    = sigmoidf(acc[1][nt][r] + vbu);
            const float ff    = sigmoidf(acc[2][nt][r] + vbf);
            const float cnew  = fu * cdash + ff * cst[idx];
            cst[idx] = cnew;
            anext[idx] = f2bf(ff * tanhf(cnew));
        }
    }
}

// -------- head: out[b][t][:] = sigmoid(a_t[b] @ W1 + b1) @ W2 + b2 ----------
__global__ void __launch_bounds__(256)
k_head(const ushort_t* __restrict__ aseq,   // [T][B][HID] bf16
       const float* __restrict__ W1, const float* __restrict__ b1,
       const float* __restrict__ W2, const float* __restrict__ b2,
       float* __restrict__ out) {
    __shared__ float s_a[HID_];
    __shared__ float s_h[64];
    const int rowid = blockIdx.x;            // t*B + b
    const int t = rowid / B_;
    const int b = rowid % B_;
    const int tid = threadIdx.x;
    const ushort_t* ar = aseq + ((size_t)t * B_ + b) * HID_;

    for (int k = tid; k < HID_; k += 256) s_a[k] = bf2f(ar[k]);
    __syncthreads();

    if (tid < 50) {
        float acc = b1[tid];
        for (int k = 0; k < HID_; ++k) acc += s_a[k] * W1[(size_t)k * 50 + tid];
        s_h[tid] = sigmoidf(acc);
    }
    __syncthreads();

    for (int o = tid; o < OUT_; o += 256) {
        float acc = b2[o];
#pragma unroll
        for (int j = 0; j < 50; ++j) acc += s_h[j] * W2[(size_t)j * OUT_ + o];
        out[((size_t)b * T_ + t) * OUT_ + o] = acc;
    }
}

// ---------------------------------------------------------------------------
extern "C" void kernel_launch(void* const* d_in, const int* in_sizes, int n_in,
                              void* d_out, int out_size, void* d_ws, size_t ws_size,
                              hipStream_t stream) {
    (void)in_sizes; (void)n_in; (void)out_size; (void)ws_size;
    const float* x  = (const float*)d_in[0];
    const float* Wu = (const float*)d_in[1];
    const float* bu = (const float*)d_in[2];
    const float* Wf = (const float*)d_in[3];
    const float* bf = (const float*)d_in[4];
    // d_in[5]=Wo, d_in[6]=bo: unused (reference bug: fo = ff)
    const float* Wh = (const float*)d_in[7];
    const float* bh = (const float*)d_in[8];
    const float* W1 = (const float*)d_in[9];
    const float* b1 = (const float*)d_in[10];
    const float* W2 = (const float*)d_in[11];
    const float* b2 = (const float*)d_in[12];
    float* out = (float*)d_out;

    // workspace layout (256B aligned chunks)
    char* p = (char*)d_ws;
    auto align256 = [](size_t v) { return (v + 255) & ~(size_t)255; };
    ushort_t* xbf = (ushort_t*)p;
    p += align256((size_t)B_ * T_ * IN_ * 2);
    ushort_t* abuf = (ushort_t*)p;                    // [T+1][B][HID]
    p += align256((size_t)(T_ + 1) * B_ * HID_ * 2);
    float* cbuf = (float*)p;
    p += align256((size_t)B_ * HID_ * 4);
    ushort_t* wt = (ushort_t*)p;                      // [3][HID][KD]

    // prep
    {
        size_t nW = (size_t)3 * HID_ * KD_;
        k_pack_w<<<(unsigned)((nW + 255) / 256), 256, 0, stream>>>(Wh, Wu, Wf, wt);
        size_t nX = (size_t)B_ * T_ * IN_;
        k_xconv<<<(unsigned)((nX + 255) / 256), 256, 0, stream>>>(x, xbf, nX);
        k_init<<<(B_ * HID_ + 255) / 256, 256, 0, stream>>>(abuf, cbuf);
    }

    // recurrence: 256 dependent steps, each a fused 3-gate WMMA GEMM + update
    const size_t slot = (size_t)B_ * HID_;
    for (int t = 0; t < T_; ++t) {
        k_step<<<32, 256, 0, stream>>>(xbf,
                                       abuf + (size_t)t * slot,
                                       abuf + (size_t)(t + 1) * slot,
                                       cbuf, wt, bh, bu, bf, t);
    }

    // head over all (t,b) rows; a_seq = slots 1..T
    k_head<<<T_ * B_, 256, 0, stream>>>(abuf + slot, W1, b1, W2, b2, out);
}
